// SDP_36636071035269
// MI455X (gfx1250) — matrix operand order repaired and checked by reading, not compiled, not run on hardware
//
#include <hip/hip_runtime.h>

typedef __bf16 bf16_t;
typedef __attribute__((ext_vector_type(16))) __bf16 v16bf;
typedef __attribute__((ext_vector_type(8)))  float  v8f;

// Problem constants
#define NB   16
#define NC   256
#define HI   128
#define WI   128
#define HWI  (HI*WI)     // 16384
#define HWP  (64*64)     // 4096 (Pi1 plane)

// LDS layout (bytes). Row strides padded for bank-conflict-free access,
// all rows 16B aligned for b128 ds loads.
#define STRIDE_A  528    // 264 halves: sCi/sPu/sQ/sK rows (256 data + 8 pad)
#define STRIDE_VT 144    // 72 halves: sVt rows (64 data + 8 pad); also sP rows
#define STRIDE_S  288    // 72 floats: sS rows
#define STRIDE_O  1032   // 258 floats: sO rows
#define SCI 0            // 64 x 528  = 33792   (dead after stage 2)
#define SPU 33792        // 64 x 528  = 33792   (dead after stage 2)
#define SQO 67584        // 64 x 528  = 33792   (dead after stage 3)
#define SKO 101376       // 64 x 528  = 33792   (dead after stage 3)
#define SVT 135168       // 256 x 144 = 36864
#define SSO 0            // alias over sCi: 64 x 288 = 18432
#define SPO 18432        // alias over sCi: 64 x 144 = 9216 (ends 27648 <= 33792)
#define SOO 67584        // alias over sQ+sK: 64 x 1032 = 66048 <= 67584
#define SMEM_BYTES 172032

union V16U { v16bf v; uint4 q[2]; };

// A-operand (M=16,K=32 bf16): lanes 0-15 take K={k0..k0+7, k0+16..k0+23},
// lanes 16-31 take the same +8. Two aligned 16B chunks per lane.
__device__ __forceinline__ v16bf loadAhalves(const char* rowBase, int k0, bool hi8) {
    const char* p = rowBase + (k0 + (hi8 ? 8 : 0)) * 2;
    V16U u;
    u.q[0] = *(const uint4*)(p);
    u.q[1] = *(const uint4*)(p + 32);
    return u.v;
}
// B-operand (K=32,N=16 bf16): lane holds one column n; lanes 0-15 K={k0..k0+7, k0+16..k0+23},
// lanes 16-31 the same +8. Two aligned 16B chunks per lane.
__device__ __forceinline__ v16bf loadBhalves(const char* rowBase, int k0, bool hi16) {
    const char* p = rowBase + (k0 + (hi16 ? 8 : 0)) * 2;
    V16U u;
    u.q[0] = *(const uint4*)(p);
    u.q[1] = *(const uint4*)(p + 32);
    return u.v;
}

__device__ __forceinline__ v8f wmma_bf16(v16bf a, v16bf b, v8f c) {
    return __builtin_amdgcn_wmma_f32_16x16x32_bf16(
        /*neg_a=*/false, a, /*neg_b=*/false, b,
        /*c_mod=*/(short)0, c, /*reuse_a=*/false, /*reuse_b=*/false);
}

extern "C" __global__ __launch_bounds__(256, 1)
void win_attn_kernel(const float* __restrict__ Ci, const float* __restrict__ Pi1,
                     const bf16_t* __restrict__ Wqkv, float* __restrict__ out) {
    extern __shared__ char smem[];
    const int tid  = threadIdx.x;
    const int blk  = blockIdx.x;
    const int b    = blk >> 8;          // 16 batches
    const int win  = blk & 255;         // 256 windows per image
    const int wy   = win >> 4, wx = win & 15;

    const int  lane = tid & 31;
    const int  lo   = lane & 15;
    const bool hiH  = lane >= 16;
    const int  wid  = tid >> 5;         // 8 waves

    // per-thread pixel / global coords (reused by stage 1 and epilogue)
    const int  p  = tid & 63;
    const int  dy = p >> 3, dx = p & 7;
    const int  Y  = wy * 8 + dy, X = wx * 8 + dx;
    const long ciB = (long)b * (NC * HWI) + (long)Y * WI + X;

    // ---------- Stage 1: stage Ci (bf16) and bilinear-upsampled Pu (bf16) ----------
    {
        // half-pixel 2x bilinear: src = out*0.5 - 0.25
        int y0 = (Y - 1) >> 1, x0 = (X - 1) >> 1;
        const float fy = (Y & 1) ? 0.25f : 0.75f;
        const float fx = (X & 1) ? 0.25f : 0.75f;
        const int yl = y0 < 0 ? 0 : y0, yh = (y0 + 1 > 63) ? 63 : y0 + 1;
        const int xl = x0 < 0 ? 0 : x0, xh = (x0 + 1 > 63) ? 63 : x0 + 1;
        const float w00 = (1.f - fy) * (1.f - fx), w01 = (1.f - fy) * fx;
        const float w10 = fy * (1.f - fx),         w11 = fy * fx;
        const long puB = (long)b * (NC * HWP);
        const int o00 = yl * 64 + xl, o01 = yl * 64 + xh;
        const int o10 = yh * 64 + xl, o11 = yh * 64 + xh;
        int c = tid >> 6;                     // 4 channel groups, +4 each iter
        for (int i = 0; i < 64; ++i, c += 4) {
            float cv = Ci[ciB + (long)c * HWI];
            *(bf16_t*)(smem + SCI + p * STRIDE_A + c * 2) = (bf16_t)cv;
            const float* pc = Pi1 + puB + (long)c * HWP;
            float pv = w00 * pc[o00] + w01 * pc[o01] + w10 * pc[o10] + w11 * pc[o11];
            *(bf16_t*)(smem + SPU + p * STRIDE_A + c * 2) = (bf16_t)pv;
        }
    }
    __syncthreads();

    // ---------- Stage 2: projections. Q = Ciw*Wq^T, K = Pu*Wk^T, V^T = (Pu*Wv^T)^T ----------
    // 192 tiles (3 mats x 4 Mtiles x 16 Ntiles), 24 per wave, 8 WMMA k-steps each.
    for (int t = wid; t < 192; t += 8) {
        const int which = t >> 6;             // 0=Q(from Ci), 1=K, 2=V (from Pu)
        const int tl = t & 63;
        const int mt = tl >> 4, nt = tl & 15;
        const char* aRow = smem + (which == 0 ? SCI : SPU) + (mt * 16 + lo) * STRIDE_A;
        const char* wRow = (const char*)(Wqkv + which * 65536 + (nt * 16 + lo) * 256);
        v8f acc = {};
#pragma unroll
        for (int kk = 0; kk < 8; ++kk) {
            v16bf a  = loadAhalves(aRow, kk * 32, hiH);
            v16bf bm = loadBhalves(wRow, kk * 32, hiH);
            acc = wmma_bf16(a, bm, acc);
        }
        if (which < 2) {                      // Q/K: row-major [pixel][ch] bf16
            char* dst = smem + (which == 0 ? SQO : SKO);
            const int o = nt * 16 + lo;
#pragma unroll
            for (int r = 0; r < 8; ++r) {
                const int pp = mt * 16 + r + (hiH ? 8 : 0);
                *(bf16_t*)(dst + pp * STRIDE_A + o * 2) = (bf16_t)acc[r];
            }
        } else {                              // V: transposed [ch][pixel] bf16, one 16B store
            union { uint4 q; bf16_t h[8]; } pk;
#pragma unroll
            for (int r = 0; r < 8; ++r) pk.h[r] = (bf16_t)acc[r];
            const int o  = nt * 16 + lo;
            const int p0 = mt * 16 + (hiH ? 8 : 0);
            *(uint4*)(smem + SVT + o * STRIDE_VT + p0 * 2) = pk.q;
        }
    }
    __syncthreads();

    // ---------- Stage 3: scores S = Q K^T / 16 ----------
    for (int st = wid; st < 16; st += 8) {
        const int mt = st >> 2, nt = st & 3;
        const char* aRow = smem + SQO + (mt * 16 + lo) * STRIDE_A;
        const char* bRow = smem + SKO + (nt * 16 + lo) * STRIDE_A;
        v8f acc = {};
#pragma unroll
        for (int kk = 0; kk < 8; ++kk) {
            v16bf a  = loadAhalves(aRow, kk * 32, hiH);
            v16bf bm = loadBhalves(bRow, kk * 32, hiH);
            acc = wmma_bf16(a, bm, acc);
        }
        const int n = nt * 16 + lo;
#pragma unroll
        for (int r = 0; r < 8; ++r) {
            const int q = mt * 16 + r + (hiH ? 8 : 0);
            *(float*)(smem + SSO + q * STRIDE_S + n * 4) = acc[r] * 0.0625f;
        }
    }
    __syncthreads();

    // ---------- Stage 4: softmax over k (64 rows, one thread each) ----------
    if (tid < 64) {
        float* row = (float*)(smem + SSO + tid * STRIDE_S);
        float mx = -3.402823466e38f;
        for (int k = 0; k < 64; ++k) mx = fmaxf(mx, row[k]);
        float s = 0.f;
        for (int k = 0; k < 64; ++k) { float e = __expf(row[k] - mx); row[k] = e; s += e; }
        const float inv = 1.f / s;
        bf16_t* prow = (bf16_t*)(smem + SPO + tid * STRIDE_VT);
        for (int k = 0; k < 64; ++k) prow[k] = (bf16_t)(row[k] * inv);
    }
    __syncthreads();

    // ---------- Stage 5: O = P * V  (A = probs 64x64 bf16, B = V^T rows) ----------
    for (int ot = wid; ot < 64; ot += 8) {
        const int mt = ot >> 4, nt = ot & 15;
        const char* aRow = smem + SPO + (mt * 16 + lo) * STRIDE_VT;
        const char* bRow = smem + SVT + (nt * 16 + lo) * STRIDE_VT;
        v8f acc = {};
#pragma unroll
        for (int kk = 0; kk < 2; ++kk) {
            v16bf a  = loadAhalves(aRow, kk * 32, hiH);
            v16bf bm = loadBhalves(bRow, kk * 32, hiH);
            acc = wmma_bf16(a, bm, acc);
        }
        const int cch = nt * 16 + lo;
#pragma unroll
        for (int r = 0; r < 8; ++r) {
            const int pp = mt * 16 + r + (hiH ? 8 : 0);
            *(float*)(smem + SOO + pp * STRIDE_O + cch * 4) = acc[r];
        }
    }
    __syncthreads();

    // ---------- Stage 6: residual epilogue, exact f32 add ----------
    {
        int c = tid >> 6;
        for (int i = 0; i < 64; ++i, c += 4) {
            const long g = ciB + (long)c * HWI;
            out[g] = Ci[g] + *(const float*)(smem + SOO + p * STRIDE_O + c * 4);
        }
    }
}

// Prep: convert f32 weights -> bf16 in workspace (B operand layout is plain row-major).
extern "C" __global__ void cvt_bf16_kernel(const float* __restrict__ src,
                                           bf16_t* __restrict__ dst, int n) {
    int i = blockIdx.x * blockDim.x + threadIdx.x;
    if (i < n) dst[i] = (bf16_t)src[i];
}

extern "C" void kernel_launch(void* const* d_in, const int* in_sizes, int n_in,
                              void* d_out, int out_size, void* d_ws, size_t ws_size,
                              hipStream_t stream) {
    const float* Ci  = (const float*)d_in[0];
    const float* Pi1 = (const float*)d_in[1];
    const float* Wq  = (const float*)d_in[2];
    const float* Wk  = (const float*)d_in[3];
    const float* Wv  = (const float*)d_in[4];
    float* out = (float*)d_out;

    bf16_t* wbf = (bf16_t*)d_ws;   // 3 x 65536 bf16 = 384 KB
    cvt_bf16_kernel<<<256, 256, 0, stream>>>(Wq, wbf,           65536);
    cvt_bf16_kernel<<<256, 256, 0, stream>>>(Wk, wbf + 65536,   65536);
    cvt_bf16_kernel<<<256, 256, 0, stream>>>(Wv, wbf + 131072,  65536);

    (void)hipFuncSetAttribute(reinterpret_cast<const void*>(win_attn_kernel),
                              hipFuncAttributeMaxDynamicSharedMemorySize, SMEM_BYTES);
    win_attn_kernel<<<dim3(NB * 256), dim3(256), SMEM_BYTES, stream>>>(Ci, Pi1, wbf, out);
}
